// GraphAttentionLayer_81235011436944
// MI455X (gfx1250) — matrix operand order repaired and checked by reading, not compiled, not run on hardware
//
#include <hip/hip_runtime.h>

#define N_NODES 8192
#define F_INDIM 512
#define F_OUTDIM 64
#define LRELU_ALPHA 0.2f

typedef __bf16 bf16_t;
typedef __attribute__((ext_vector_type(16))) __bf16 v16bf;
typedef __attribute__((ext_vector_type(8)))  float  v8f;
typedef unsigned int u32x4 __attribute__((ext_vector_type(4)));
typedef int          i32x8 __attribute__((ext_vector_type(8)));
typedef int          i32x4 __attribute__((ext_vector_type(4)));

// Native convert: backend emits v_cvt_pk_bf16_f32.
static __device__ __forceinline__ bf16_t f2bf(float f) { return (bf16_t)f; }

static __device__ __forceinline__ float lrelu(float x) {
  return x > 0.f ? x : LRELU_ALPHA * x;
}

// Byte offset of a (generic) pointer inside the workgroup LDS allocation.
static __device__ __forceinline__ unsigned lds_addr_of(const void* p) {
  return (unsigned)(unsigned long long)(__attribute__((address_space(3))) const void*)p;
}

// TDM: DMA one 16x32 tile of i32 from a 8192x8192 tensor into LDS.
// D# per CDNA5 ISA ch. 8: group0 = count/lds_addr/global_addr/type,
// group1 = data_size, tensor dims, tile dims, stride; groups 2/3 zero (2D).
// This toolchain exposes the 6-arg builtin (extra i32x8 group before cpol).
static __device__ __forceinline__ void tdm_load_adj_tile(unsigned lds_byte_off,
                                                         const int* gtile) {
  unsigned long long ga = (unsigned long long)(const void*)gtile;
  u32x4 g0;
  g0.x = 1u;                                   // count=1 (valid user descriptor)
  g0.y = lds_byte_off;                         // LDS destination byte address
  g0.z = (unsigned)(ga & 0xFFFFFFFFu);         // global_addr[31:0]   (bits 95:64)
  g0.w = (unsigned)(ga >> 32) | 0x80000000u;   // global_addr[56:32], type=2 @127:126
  i32x8 g1;
  g1[0] = 0x00020000;                          // wg_mask=0, data_size=2 (4 bytes)
  g1[1] = (int)((N_NODES & 0xFFFFu) << 16);    // tensor_dim0[15:0] @ bits 63:48
  g1[2] = (int)(((unsigned)N_NODES >> 16) | ((N_NODES & 0xFFFFu) << 16)); // dim0 hi | dim1 lo
  g1[3] = (int)(((unsigned)N_NODES >> 16) | (32u << 16));                 // dim1 hi | tile_dim0=32
  g1[4] = 16;                                  // tile_dim1=16, tile_dim2=0
  g1[5] = N_NODES;                             // tensor_dim0_stride[31:0] = 8192
  g1[6] = 0;                                   // stride0 hi | stride1 lo
  g1[7] = 0;
  i32x4 z4 = {0, 0, 0, 0};
  i32x8 z8 = {0, 0, 0, 0, 0, 0, 0, 0};
  __builtin_amdgcn_tensor_load_to_lds(g0, g1, z4, z4, z8, 0);
}

// ---------------- Kernel 0: W (512x64 f32) -> WT (64x512 bf16) ----------------
__global__ void k0_transpose_w(const float* __restrict__ W, bf16_t* __restrict__ WT) {
  int idx = blockIdx.x * blockDim.x + threadIdx.x;     // 0..32767
  if (idx < F_INDIM * F_OUTDIM) {
    int k = idx / F_OUTDIM;
    int n = idx % F_OUTDIM;
    WT[n * F_INDIM + k] = f2bf(W[idx]);
  }
}

// ---------------- Kernel 1: h = X @ W (bf16 WMMA), store hbT (64x8192 bf16),
//                  s_src = h @ a[:64], s_dst = h @ a[64:] ----------------
__global__ void __launch_bounds__(256)
k1_gemm_h(const float* __restrict__ X,
          const bf16_t* __restrict__ WT,
          const float* __restrict__ avec,
          bf16_t* __restrict__ hbT,
          float* __restrict__ s_src,
          float* __restrict__ s_dst) {
  const int lane  = threadIdx.x & 31;
  const int wid   = threadIdx.x >> 5;
  const int tile  = blockIdx.x * 8 + wid;     // 512 tiles of 16 rows
  const int i0    = tile * 16;
  const int row   = lane & 15;
  const int half  = lane >> 4;                // 0 or 1
  const int khalf = half * 8;

  v8f c[4] = {};
  const float* xrow = X + (size_t)(i0 + row) * F_INDIM;
  const bf16_t* wcol = WT + (size_t)row * F_INDIM;

  for (int k0 = 0; k0 < F_INDIM; k0 += 32) {
    const bf16_t* bp = wcol + k0 + half * 16;
    const v16bf b0 = *(const v16bf*)(bp);
    const v16bf b1 = *(const v16bf*)(bp + (size_t)16 * F_INDIM);
    const v16bf b2 = *(const v16bf*)(bp + (size_t)32 * F_INDIM);
    const v16bf b3 = *(const v16bf*)(bp + (size_t)48 * F_INDIM);

    // A: 16x32 bf16, contiguous K order (lane<16: K{0..15}; lane>=16: K{16..31})
    float xv[16];
    *(float4*)(xv + 0)  = *(const float4*)(xrow + k0 + half * 16);
    *(float4*)(xv + 4)  = *(const float4*)(xrow + k0 + half * 16 + 4);
    *(float4*)(xv + 8)  = *(const float4*)(xrow + k0 + half * 16 + 8);
    *(float4*)(xv + 12) = *(const float4*)(xrow + k0 + half * 16 + 12);
    v16bf a;
#pragma unroll
    for (int t = 0; t < 16; ++t) a[t] = f2bf(xv[t]);

    c[0] = __builtin_amdgcn_wmma_f32_16x16x32_bf16(false, a, false, b0, (short)0, c[0], false, false);
    c[1] = __builtin_amdgcn_wmma_f32_16x16x32_bf16(false, a, false, b1, (short)0, c[1], false, false);
    c[2] = __builtin_amdgcn_wmma_f32_16x16x32_bf16(false, a, false, b2, (short)0, c[2], false, false);
    c[3] = __builtin_amdgcn_wmma_f32_16x16x32_bf16(false, a, false, b3, (short)0, c[3], false, false);
  }

  // Epilogue: C layout -> VGPR r holds (M = half*8 + r, col = lane&15) per n-tile
  const int mbase = half * 8;
  float psrc[8] = {}, pdst[8] = {};
#pragma unroll
  for (int nt = 0; nt < 4; ++nt) {
    const int gcol = nt * 16 + row;
    const float as = avec[gcol];
    const float ad = avec[F_OUTDIM + gcol];
#pragma unroll
    for (int r = 0; r < 8; ++r) {
      const float v = c[nt][r];
      hbT[(size_t)gcol * N_NODES + (i0 + mbase + r)] = f2bf(v);
      psrc[r] += v * as;
      pdst[r] += v * ad;
    }
  }
#pragma unroll
  for (int m = 1; m <= 8; m <<= 1) {
#pragma unroll
    for (int r = 0; r < 8; ++r) {
      psrc[r] += __shfl_xor(psrc[r], m, 32);
      pdst[r] += __shfl_xor(pdst[r], m, 32);
    }
  }
  if (row == 0) {
#pragma unroll
    for (int r = 0; r < 8; ++r) {
      s_src[i0 + mbase + r] = psrc[r];
      s_dst[i0 + mbase + r] = pdst[r];
    }
  }
}

// ---------------- Kernel 2: smax = max(s_dst) ----------------
__global__ void k2_max(const float* __restrict__ s_dst, float* __restrict__ smax) {
  __shared__ float red[256];
  float m = -3.0e38f;
  for (int i = threadIdx.x; i < N_NODES; i += 256) m = fmaxf(m, s_dst[i]);
  red[threadIdx.x] = m;
  __syncthreads();
  for (int s = 128; s > 0; s >>= 1) {
    if ((int)threadIdx.x < s) red[threadIdx.x] = fmaxf(red[threadIdx.x], red[threadIdx.x + s]);
    __syncthreads();
  }
  if (threadIdx.x == 0) *smax = red[0];
}

// ---------------- Kernel 3: fused masked-softmax + attention@h + ELU ----------------
// Single pass: stabilizer m_i = lrelu(s_src[i] + max(s_dst)) bounds every row element
// (leaky_relu monotone), so no online-max rescaling is needed. Adjacency tiles are
// streamed into LDS by the Tensor Data Mover, double-buffered per wave.
__global__ void __launch_bounds__(256)
k3_attn(const int* __restrict__ adj,
        const bf16_t* __restrict__ hbT,
        const float* __restrict__ s_src,
        const float* __restrict__ s_dst,
        const float* __restrict__ smaxp,
        float* __restrict__ out) {
  __shared__ int   adjbuf[2][8][16 * 32];   // 32 KB: TDM double-buffer, per wave
  __shared__ float sdsh[N_NODES];           // 32 KB

  for (int i = threadIdx.x; i < N_NODES; i += 256) sdsh[i] = s_dst[i];
  __syncthreads();

  const int lane  = threadIdx.x & 31;
  const int wid   = threadIdx.x >> 5;
  const int tile  = blockIdx.x * 8 + wid;   // 512 tiles of 16 rows
  const int i0    = tile * 16;
  const int row   = lane & 15;
  const int half  = lane >> 4;
  const int khalf = half * 8;

  const float smax = *smaxp;
  const float ssr  = s_src[i0 + row];
  const float mrow = lrelu(ssr + smax);     // per-row softmax stabilizer (upper bound)

  const int* gadj0 = adj + (size_t)i0 * N_NODES;   // top-left of this wave's row band

  v8f   c[4] = {};
  float lsum = 0.f;

  // Prologue: TDM fetch of the first tile.
  tdm_load_adj_tile(lds_addr_of(&adjbuf[0][wid][0]), gadj0);

  for (int j0 = 0; j0 < N_NODES; j0 += 32) {
    const int buf = (j0 >> 5) & 1;
    if (j0 + 32 < N_NODES) {
      tdm_load_adj_tile(lds_addr_of(&adjbuf[buf ^ 1][wid][0]), gadj0 + (j0 + 32));
      __builtin_amdgcn_s_wait_tensorcnt(1);   // current buffer complete (in-order)
    } else {
      __builtin_amdgcn_s_wait_tensorcnt(0);
    }
    asm volatile("" ::: "memory");            // keep LDS reads after the wait

    // B tiles (hbT is 64 x 8192 bf16, K contiguous).
    const int kb = j0 + half * 16;
    const bf16_t* bp = hbT + (size_t)row * N_NODES + kb;
    const v16bf b0 = *(const v16bf*)(bp);
    const v16bf b1 = *(const v16bf*)(bp + (size_t)16 * N_NODES);
    const v16bf b2 = *(const v16bf*)(bp + (size_t)32 * N_NODES);
    const v16bf b3 = *(const v16bf*)(bp + (size_t)48 * N_NODES);

    // Adjacency tile from LDS (row-major 16x32 as written by TDM).
    const int* tb = &adjbuf[buf][wid][0];
    int aj[16];
    *(int4*)(aj + 0)  = *(const int4*)(tb + row * 32 + half * 16);
    *(int4*)(aj + 4)  = *(const int4*)(tb + row * 32 + half * 16 + 4);
    *(int4*)(aj + 8)  = *(const int4*)(tb + row * 32 + half * 16 + 8);
    *(int4*)(aj + 12) = *(const int4*)(tb + row * 32 + half * 16 + 12);

    // s_dst from LDS.
    float sdv[16];
    *(float4*)(sdv + 0)  = *(const float4*)(&sdsh[j0 + half * 16]);
    *(float4*)(sdv + 4)  = *(const float4*)(&sdsh[j0 + half * 16 + 4]);
    *(float4*)(sdv + 8)  = *(const float4*)(&sdsh[j0 + half * 16 + 8]);
    *(float4*)(sdv + 12) = *(const float4*)(&sdsh[j0 + half * 16 + 12]);

    v16bf a;
#pragma unroll
    for (int t = 0; t < 16; ++t) {
      const float e = lrelu(ssr + sdv[t]);
      const float p = (aj[t] > 0) ? __expf(e - mrow) : 0.f;
      lsum += p;
      a[t] = f2bf(p);
    }

    c[0] = __builtin_amdgcn_wmma_f32_16x16x32_bf16(false, a, false, b0, (short)0, c[0], false, false);
    c[1] = __builtin_amdgcn_wmma_f32_16x16x32_bf16(false, a, false, b1, (short)0, c[1], false, false);
    c[2] = __builtin_amdgcn_wmma_f32_16x16x32_bf16(false, a, false, b2, (short)0, c[2], false, false);
    c[3] = __builtin_amdgcn_wmma_f32_16x16x32_bf16(false, a, false, b3, (short)0, c[3], false, false);
  }

  // Full row sums: row owned by lanes L and L+16
  lsum += __shfl_xor(lsum, 16, 32);

  const int mbase = half * 8;
  float lr[8];
#pragma unroll
  for (int r = 0; r < 8; ++r) lr[r] = __shfl(lsum, mbase + r, 32);

#pragma unroll
  for (int nt = 0; nt < 4; ++nt) {
    const int gcol = nt * 16 + row;
#pragma unroll
    for (int r = 0; r < 8; ++r) {
      const float hp = c[nt][r] / lr[r];
      const float o  = hp > 0.f ? hp : (__expf(hp) - 1.f);   // ELU(alpha=1)
      out[(size_t)(i0 + mbase + r) * F_OUTDIM + gcol] = o;
    }
  }
}

extern "C" void kernel_launch(void* const* d_in, const int* in_sizes, int n_in,
                              void* d_out, int out_size, void* d_ws, size_t ws_size,
                              hipStream_t stream) {
  (void)in_sizes; (void)n_in; (void)out_size; (void)ws_size;

  const float* X   = (const float*)d_in[0];   // (8192, 512) f32
  const int*   adj = (const int*)d_in[1];     // (8192, 8192) i32
  const float* W   = (const float*)d_in[2];   // (512, 64) f32
  const float* a   = (const float*)d_in[3];   // (128, 1) f32
  float*       out = (float*)d_out;           // (8192, 64) f32

  char* ws = (char*)d_ws;
  bf16_t* WT    = (bf16_t*)ws;                                  // 64*512*2   = 64 KB
  bf16_t* hbT   = (bf16_t*)(ws + 65536);                        // 64*8192*2  = 1 MB
  float*  s_src = (float*)(ws + 65536 + 1048576);               // 32 KB
  float*  s_dst = s_src + N_NODES;                              // 32 KB
  float*  smax  = s_dst + N_NODES;                              // 4 B

  k0_transpose_w<<<128, 256, 0, stream>>>(W, WT);
  k1_gemm_h     <<<64, 256, 0, stream>>>(X, WT, a, hbT, s_src, s_dst);
  k2_max        <<<1, 256, 0, stream>>>(s_dst, smax);
  k3_attn       <<<64, 256, 0, stream>>>(adj, hbT, s_src, s_dst, smax, out);
}
